// TurboQuantProd_876173328528
// MI455X (gfx1250) — matrix operand order repaired and checked by reading, not compiled, not run on hardware
//
#include <hip/hip_runtime.h>
#include <hip/hip_bf16.h>
#include <math.h>

typedef __attribute__((ext_vector_type(16))) _Float16 v16h;
typedef __attribute__((ext_vector_type(8)))  float    v8f;

#define DDIM  1024
#define BATCH 32768
#define MBLK  64
#define PITCH 1032   // ushorts per LDS row (2064 B) -> bank de-conflict padding

union FragU { uint4 q[2]; v16h h; };

__device__ __forceinline__ unsigned short f2h_bits(float f) {
  union { _Float16 h; unsigned short u; } c;
  c.h = (_Float16)f;
  return c.u;
}

// B fragment (32x16 f16): lane = column n0+(lane&15); k-chunks [kb,kb+8) and
// [kb+16,kb+24), kb = k0 + (lane>=16 ? 8 : 0). Memory layout: mat[n*1024 + k] (k contiguous).
__device__ __forceinline__ v16h load_b_frag(const unsigned short* __restrict__ mat,
                                            int n0, int k0, int lane) {
  const unsigned short* p = mat + (unsigned)(n0 + (lane & 15)) * DDIM
                                + k0 + ((lane & 16) ? 8 : 0);
  FragU f;
  f.q[0] = ((const uint4*)p)[0];
  f.q[1] = ((const uint4*)(p + 16))[0];
  return f.h;
}

// A fragment (16x32 f16) from row-major f32 global: lane = row (lane&15),
// k-chunks [kb,kb+8) and [kb+16,kb+24), kb = k0 + (lane>=16 ? 8 : 0).
__device__ __forceinline__ v16h load_a_global_f32(const float* __restrict__ xrow0,
                                                  int k0, int lane) {
  const float* p = xrow0 + (size_t)(lane & 15) * DDIM + k0 + ((lane & 16) ? 8 : 0);
  float4 f0 = ((const float4*)p)[0];
  float4 f1 = ((const float4*)p)[1];
  float4 f2 = ((const float4*)(p + 16))[0];
  float4 f3 = ((const float4*)(p + 16))[1];
  v16h a;
  a[0]=(_Float16)f0.x;  a[1]=(_Float16)f0.y;  a[2]=(_Float16)f0.z;  a[3]=(_Float16)f0.w;
  a[4]=(_Float16)f1.x;  a[5]=(_Float16)f1.y;  a[6]=(_Float16)f1.z;  a[7]=(_Float16)f1.w;
  a[8]=(_Float16)f2.x;  a[9]=(_Float16)f2.y;  a[10]=(_Float16)f2.z; a[11]=(_Float16)f2.w;
  a[12]=(_Float16)f3.x; a[13]=(_Float16)f3.y; a[14]=(_Float16)f3.z; a[15]=(_Float16)f3.w;
  return a;
}

// A fragment (16x32 f16) from LDS f16 rows (pitch PITCH ushorts).
__device__ __forceinline__ v16h load_a_lds(const unsigned short* row0, int k0, int lane) {
  const unsigned short* p = row0 + (unsigned)(lane & 15) * PITCH
                                 + k0 + ((lane & 16) ? 8 : 0);
  FragU f;
  f.q[0] = *(const uint4*)p;
  f.q[1] = *(const uint4*)(p + 16);
  return f.h;
}

// ---------------- prep: f32 -> f16 panels (Pi, Pi^T, S) ----------------
__global__ void tq_prep(const float* __restrict__ Pi, const float* __restrict__ S,
                        unsigned short* __restrict__ PiB,
                        unsigned short* __restrict__ PiTB,
                        unsigned short* __restrict__ SB) {
  const int t = blockIdx.x * blockDim.x + threadIdx.x;   // 0 .. 1M-1
  const int n = t >> 10;
  const int k = t & (DDIM - 1);
  PiB[t]  = f2h_bits(Pi[t]);            // B for stage1: B(k,n) = Pi[n][k]
  SB[t]   = f2h_bits(S[t]);             // B for stage3: B(k,n) = S[n][k]
  PiTB[t] = f2h_bits(Pi[k * DDIM + n]); // B for stage2: B(k,n) = Pi[k][n]
}

// ---------------- fused 3-stage pipeline, 64 rows per workgroup ----------------
__global__ __launch_bounds__(256, 1)
void tq_fused(const float* __restrict__ x,
              const float* __restrict__ cen,
              const unsigned short* __restrict__ PiB,
              const unsigned short* __restrict__ PiTB,
              const unsigned short* __restrict__ SB,
              float* __restrict__ out_idx,
              float* __restrict__ out_sign,
              float* __restrict__ out_norm) {
  extern __shared__ unsigned char smem_raw[];
  unsigned short* yhat = (unsigned short*)smem_raw;      // MBLK * PITCH f16
  unsigned short* resd = yhat + MBLK * PITCH;            // MBLK * PITCH f16
  float* rowsum        = (float*)(resd + MBLK * PITCH);  // MBLK f32

  const int tid   = threadIdx.x;
  const int lane  = tid & 31;
  const int wave  = tid >> 5;
  const int mg    = wave & 3;              // row group (16 rows)
  const int nh    = wave >> 2;             // N half (512 cols)
  const int lrow0 = mg * 16;
  const size_t row0 = (size_t)blockIdx.x * MBLK;
  const int laneN = lane & 15;
  const int erow  = (lane & 16) ? 8 : 0;

  if (tid < MBLK) rowsum[tid] = 0.f;

  // Centroid constants (sorted ascending Gaussian quantiles).
  float c[8];
#pragma unroll
  for (int i = 0; i < 8; ++i) c[i] = cen[i];
  float mid[7], del[7];
#pragma unroll
  for (int j = 0; j < 7; ++j) { mid[j] = 0.5f * (c[j] + c[j + 1]); del[j] = c[j + 1] - c[j]; }

  // ---- Stage 1: y = x @ Pi^T ; quantize -> indices (global) + yhat (LDS) ----
  const float* xrow0 = x + (row0 + (size_t)lrow0) * DDIM;
  for (int nc = 0; nc < 4; ++nc) {
    const int n0 = nh * 512 + nc * 128;
    v8f acc[8] = {};
    for (int k0 = 0; k0 < DDIM; k0 += 32) {
      v16h a = load_a_global_f32(xrow0, k0, lane);
#pragma unroll
      for (int t = 0; t < 8; ++t) {
        v16h b = load_b_frag(PiB, n0 + t * 16, k0, lane);
        acc[t] = __builtin_amdgcn_wmma_f32_16x16x32_f16(false, a, false, b,
                                                        (short)0, acc[t], false, false);
      }
    }
#pragma unroll
    for (int t = 0; t < 8; ++t) {
      const int col = n0 + t * 16 + laneN;
#pragma unroll
      for (int e = 0; e < 8; ++e) {
        const int lr = lrow0 + e + erow;
        const float y = acc[t][e];
        float yh = c[0];
        int   idx = 0;
#pragma unroll
        for (int j = 0; j < 7; ++j) {
          const bool gt = y > mid[j];
          idx += gt ? 1 : 0;
          yh  += gt ? del[j] : 0.f;
        }
        out_idx[(row0 + (size_t)lr) * DDIM + col] = (float)idx;
        yhat[lr * PITCH + col] = f2h_bits(yh);
      }
    }
  }
  __syncthreads();

  // ---- Stage 2: x_hat = yhat @ Pi ; residual -> LDS ; row |r|^2 ----
  float rs[8] = {0.f, 0.f, 0.f, 0.f, 0.f, 0.f, 0.f, 0.f};
  const unsigned short* arow_y = yhat + lrow0 * PITCH;
  for (int nc = 0; nc < 4; ++nc) {
    const int n0 = nh * 512 + nc * 128;
    v8f acc[8] = {};
    for (int k0 = 0; k0 < DDIM; k0 += 32) {
      v16h a = load_a_lds(arow_y, k0, lane);
#pragma unroll
      for (int t = 0; t < 8; ++t) {
        v16h b = load_b_frag(PiTB, n0 + t * 16, k0, lane);
        acc[t] = __builtin_amdgcn_wmma_f32_16x16x32_f16(false, a, false, b,
                                                        (short)0, acc[t], false, false);
      }
    }
#pragma unroll
    for (int t = 0; t < 8; ++t) {
      const int col = n0 + t * 16 + laneN;
#pragma unroll
      for (int e = 0; e < 8; ++e) {
        const int lr = lrow0 + e + erow;
        const float xv = x[(row0 + (size_t)lr) * DDIM + col];
        const float r  = xv - acc[t][e];
        rs[e] += r * r;
        resd[lr * PITCH + col] = f2h_bits(r);
      }
    }
  }
#pragma unroll
  for (int e = 0; e < 8; ++e) atomicAdd(&rowsum[lrow0 + e + erow], rs[e]);
  __syncthreads();

  // ---- Stage 3: proj = residual @ S^T ; signs ; norms ----
  if (tid < MBLK) out_norm[row0 + tid] = sqrtf(rowsum[tid]);

  const unsigned short* arow_r = resd + lrow0 * PITCH;
  for (int nc = 0; nc < 4; ++nc) {
    const int n0 = nh * 512 + nc * 128;
    v8f acc[8] = {};
    for (int k0 = 0; k0 < DDIM; k0 += 32) {
      v16h a = load_a_lds(arow_r, k0, lane);
#pragma unroll
      for (int t = 0; t < 8; ++t) {
        v16h b = load_b_frag(SB, n0 + t * 16, k0, lane);
        acc[t] = __builtin_amdgcn_wmma_f32_16x16x32_f16(false, a, false, b,
                                                        (short)0, acc[t], false, false);
      }
    }
#pragma unroll
    for (int t = 0; t < 8; ++t) {
      const int col = n0 + t * 16 + laneN;
#pragma unroll
      for (int e = 0; e < 8; ++e) {
        const int lr = lrow0 + e + erow;
        out_sign[(row0 + (size_t)lr) * DDIM + col] = (acc[t][e] >= 0.f) ? 1.f : -1.f;
      }
    }
  }
}

extern "C" void kernel_launch(void* const* d_in, const int* in_sizes, int n_in,
                              void* d_out, int out_size, void* d_ws, size_t ws_size,
                              hipStream_t stream) {
  const float* x   = (const float*)d_in[0];   // (32768, 1024)
  const float* Pi  = (const float*)d_in[1];   // (1024, 1024)
  const float* cen = (const float*)d_in[2];   // (8,)
  const float* S   = (const float*)d_in[3];   // (1024, 1024)

  unsigned short* PiB  = (unsigned short*)d_ws;          // 2 MB
  unsigned short* PiTB = PiB  + (size_t)DDIM * DDIM;     // 2 MB
  unsigned short* SB   = PiTB + (size_t)DDIM * DDIM;     // 2 MB

  float* out_idx  = (float*)d_out;
  float* out_sign = out_idx  + (size_t)BATCH * DDIM;
  float* out_norm = out_sign + (size_t)BATCH * DDIM;

  tq_prep<<<(DDIM * DDIM) / 256, 256, 0, stream>>>(Pi, S, PiB, PiTB, SB);

  const size_t shmem = (size_t)2 * MBLK * PITCH * sizeof(unsigned short)
                     + (size_t)MBLK * sizeof(float);     // ~258.3 KB
  tq_fused<<<BATCH / MBLK, 256, shmem, stream>>>(x, cen, PiB, PiTB, SB,
                                                 out_idx, out_sign, out_norm);
}
